// WindowSA_644245094964
// MI455X (gfx1250) — matrix operand order repaired and checked by reading, not compiled, not run on hardware
//
#include <hip/hip_runtime.h>
#include <hip/hip_bf16.h>
#include <math.h>

// ---------------- problem constants ----------------
#define DIM    384
#define DH     32
#define NHEAD  12
#define NWIN   3072
#define WACT   1536
#define TT     64
#define NTOK   (WACT * TT)      // 98304
#define NBLOCK 9830
#define MLPD   (4 * DIM)        // 1536
#define SEQ    64               // S = NTOK / M

typedef __attribute__((ext_vector_type(16))) __bf16 v16bf;
typedef __attribute__((ext_vector_type(8)))  __bf16 v8bf;
typedef __attribute__((ext_vector_type(8)))  float  v8f;
typedef unsigned int u32x4 __attribute__((ext_vector_type(4)));
typedef int          i32x8 __attribute__((ext_vector_type(8)));
typedef int          i32x4 __attribute__((ext_vector_type(4)));

#define HAVE_TDM __has_builtin(__builtin_amdgcn_tensor_load_to_lds)

// ---------------- helpers ----------------
__device__ __forceinline__ __bf16 f2bf(float f) {
    unsigned u = __builtin_bit_cast(unsigned, f);
    u += 0x7FFFu + ((u >> 16) & 1u);              // round-to-nearest-even
    unsigned short h = (unsigned short)(u >> 16);
    return __builtin_bit_cast(__bf16, h);
}

// 16-bit A-matrix fragment (16x32, MxK), contiguous k order (matches B loads):
// lane row = lane&15; lanes<16 hold K {0..15}, lanes>=16 hold K {16..31}
__device__ __forceinline__ v16bf load_a_frag(const __bf16* rowptr, int hi) {
    return *(const v16bf*)(rowptr + hi * 16);
}

__device__ __forceinline__ float wred_max16(float v) {
    v = fmaxf(v, __shfl_xor(v, 1));
    v = fmaxf(v, __shfl_xor(v, 2));
    v = fmaxf(v, __shfl_xor(v, 4));
    v = fmaxf(v, __shfl_xor(v, 8));
    return v;
}
__device__ __forceinline__ float wred_sum16(float v) {
    v += __shfl_xor(v, 1);
    v += __shfl_xor(v, 2);
    v += __shfl_xor(v, 4);
    v += __shfl_xor(v, 8);
    return v;
}

__device__ __forceinline__ void wait_tensor0() {
#if __has_builtin(__builtin_amdgcn_s_wait_tensorcnt)
    __builtin_amdgcn_s_wait_tensorcnt(0);
#else
    asm volatile("s_wait_tensorcnt 0" ::: "memory");
#endif
}

#if HAVE_TDM
// Issue a TDM 2D tile load: tile_dim0=32 elems (k), tile_dim1=64 rows, 2-byte data.
// D# bit layout per CDNA5 ISA ch.8 (groups 0/1; groups 2/3 zero => 2D tensor).
__device__ __forceinline__ void tdm_load_tile_2d(unsigned lds_addr, const __bf16* gaddr,
                                                 unsigned tensor_d0, unsigned tensor_d1,
                                                 unsigned long long stride_d0) {
    unsigned long long ga = (unsigned long long)(__UINTPTR_TYPE__)gaddr;
    u32x4 g0;
    g0[0] = 1u;                                               // count=1 (valid user D#)
    g0[1] = lds_addr;                                         // lds_addr [63:32]
    g0[2] = (unsigned)(ga & 0xFFFFFFFFu);                     // global_addr[31:0]
    g0[3] = (unsigned)((ga >> 32) & 0x01FFFFFFu) | (2u << 30);// global_addr[56:32], type=2
    i32x8 g1;
    g1[0] = (int)(1u << 16);                                  // data_size=1 -> 2 bytes
    g1[1] = (int)((tensor_d0 & 0xFFFFu) << 16);               // tensor_dim0[15:0]
    g1[2] = (int)(((tensor_d0 >> 16) & 0xFFFFu) |
                  ((tensor_d1 & 0xFFFFu) << 16));             // td0[31:16] | td1[15:0]
    g1[3] = (int)(((tensor_d1 >> 16) & 0xFFFFu) | (32u << 16));// td1[31:16] | tile_dim0=32
    g1[4] = (int)64u;                                         // tile_dim1=64, tile_dim2=0
    g1[5] = (int)(unsigned)(stride_d0 & 0xFFFFFFFFull);       // tensor_dim0_stride[31:0]
    g1[6] = (int)(unsigned)((stride_d0 >> 32) & 0xFFFFull);   // stride[47:32] | d1_stride[15:0]=0
    g1[7] = 0;
    i32x4 z4 = {0, 0, 0, 0};
#if __has_include(<hip/amd_detail/amd_gfx1250_TDM.h>)
    i32x8 z8 = {0, 0, 0, 0, 0, 0, 0, 0};
    __builtin_amdgcn_tensor_load_to_lds(g0, g1, z4, z4, z8, 0);
#else
    __builtin_amdgcn_tensor_load_to_lds(g0, g1, z4, z4, 0);
#endif
}
#endif

// ---------------- LayerNorm (one wave per 384-wide row) ----------------
template<bool BF16OUT>
__global__ __launch_bounds__(256) void ln_kernel(const float* __restrict__ x,
                                                 const float* __restrict__ w,
                                                 const float* __restrict__ b,
                                                 float* __restrict__ outF,
                                                 __bf16* __restrict__ outB,
                                                 int rows) {
    int wid  = (int)((blockIdx.x * 256 + threadIdx.x) >> 5);
    int lane = threadIdx.x & 31;
    if (wid >= rows) return;
    const float* xr = x + (size_t)wid * DIM;
    float v[12];
    float s = 0.f;
#pragma unroll
    for (int i = 0; i < 12; ++i) { v[i] = xr[lane + i * 32]; s += v[i]; }
    s += __shfl_xor(s, 1);  s += __shfl_xor(s, 2);
    s += __shfl_xor(s, 4);  s += __shfl_xor(s, 8); s += __shfl_xor(s, 16);
    float mu = s * (1.0f / DIM);
    float s2 = 0.f;
#pragma unroll
    for (int i = 0; i < 12; ++i) { float d = v[i] - mu; s2 += d * d; }
    s2 += __shfl_xor(s2, 1); s2 += __shfl_xor(s2, 2);
    s2 += __shfl_xor(s2, 4); s2 += __shfl_xor(s2, 8); s2 += __shfl_xor(s2, 16);
    float rstd = rsqrtf(s2 * (1.0f / DIM) + 1e-5f);
#pragma unroll
    for (int i = 0; i < 12; ++i) {
        int c = lane + i * 32;
        float o = (v[i] - mu) * rstd * w[c] + b[c];
        if (BF16OUT) outB[(size_t)wid * DIM + c] = f2bf(o);
        else         outF[(size_t)wid * DIM + c] = o;
    }
}

// ---------------- misc small kernels ----------------
__global__ __launch_bounds__(256) void cast_w_kernel(const float* __restrict__ s,
                                                     __bf16* __restrict__ d, int n) {
    int i = blockIdx.x * 256 + threadIdx.x;
    if (i < n) d[i] = f2bf(s[i]);
}

__global__ __launch_bounds__(256) void scatter_flag_kernel(const int* __restrict__ bi,
                                                           unsigned char* __restrict__ flag) {
    int i = blockIdx.x * 256 + threadIdx.x;
    if (i < NBLOCK) flag[bi[i]] = 1;
}

// xp[p,c] = bf16(xn[rowOf(p), c]);  maskF[p] = blocked-key selector;  rowIdx[p] = dest row
__global__ __launch_bounds__(256) void gather_kernel(const float* __restrict__ xn,
                                                     const int* __restrict__ iw,
                                                     const int* __restrict__ ip,
                                                     const unsigned char* __restrict__ flag,
                                                     __bf16* __restrict__ xp,
                                                     float* __restrict__ maskF,
                                                     int* __restrict__ rowIdx) {
    size_t idx = (size_t)blockIdx.x * 256 + threadIdx.x;
    if (idx >= (size_t)NTOK * DIM) return;
    int p = (int)(idx / DIM), c = (int)(idx % DIM);
    int g = ip[p];
    int srow = iw[g >> 6] * TT + (g & 63);
    xp[idx] = f2bf(xn[(size_t)srow * DIM + c]);
    if (c == 0) {
        maskF[p]  = flag[g] ? 1.0f : 0.0f;
        rowIdx[p] = srow;
    }
}

// ---------------- generic WMMA GEMM: C = A(bf16, MxK) * W(bf16, NxK)^T + bias ----------------
// block = 4 waves; wave = one 16x64 output strip; grid = (Mrows/64, N/64)
// Weight chunk (64 rows x 32 k, 4 KB) staged in LDS by the Tensor Data Mover,
// double-buffered: TDM for chunk k+1 overlaps WMMAs on chunk k (TENSORcnt sync).
// EPI 0: store bf16           EPI 1: GELU(exact) -> bf16
// EPI 2: + xn shortcut (gathered via rowIdx from d_out) -> fp32 h
// EPI 3: + h residual, conditional scatter to d_out rows (skip blocked)
template<int EPI>
__global__ __launch_bounds__(128) void gemm_kernel(const __bf16* __restrict__ A,
                                                   const __bf16* __restrict__ W,
                                                   const float* __restrict__ bias,
                                                   int N, int Kdim,
                                                   __bf16* __restrict__ outB,
                                                   float* __restrict__ outF,
                                                   const float* __restrict__ resid,
                                                   const int* __restrict__ rowIdx,
                                                   const unsigned char* __restrict__ flag,
                                                   const int* __restrict__ part,
                                                   float* __restrict__ gout) {
    __shared__ __bf16 Bst[2][64 * 32];            // two 4 KB weight-chunk buffers
    const int tid  = threadIdx.x;
    const int wave = tid >> 5;
    const int lane = tid & 31;
    const int ln   = lane & 15;
    const int hi   = lane >> 4;
    const int r0   = blockIdx.x * 64 + wave * 16;
    const int c0   = blockIdx.y * 64;
    const int nk   = Kdim / 32;

    const __bf16* arow = A + (size_t)(r0 + ln) * Kdim;
    v8f acc[4] = {};

#if HAVE_TDM
    const unsigned lds0 = (unsigned)(__UINTPTR_TYPE__)&Bst[0][0];
    const unsigned lds1 = (unsigned)(__UINTPTR_TYPE__)&Bst[1][0];
    if (tid < 32)                                  // wave 0 issues DMA (EXEC ignored by TDM)
        tdm_load_tile_2d(lds0, W + (size_t)c0 * Kdim, (unsigned)Kdim, (unsigned)N,
                         (unsigned long long)Kdim);
    for (int ki = 0; ki < nk; ++ki) {
        if (tid < 32) wait_tensor0();              // chunk ki landed in LDS
        __syncthreads();                           // all waves see it; prev buffer free
        if ((tid < 32) && (ki + 1 < nk))           // prefetch chunk ki+1 into other buffer
            tdm_load_tile_2d((ki & 1) ? lds0 : lds1,
                             W + (size_t)c0 * Kdim + (size_t)(ki + 1) * 32,
                             (unsigned)Kdim, (unsigned)N, (unsigned long long)Kdim);
        const __bf16* bbase = &Bst[ki & 1][0];
        v16bf af = load_a_frag(arow + ki * 32, hi);
#pragma unroll
        for (int j = 0; j < 4; ++j) {
            v16bf bf = *(const v16bf*)(bbase + (j * 16 + ln) * 32 + hi * 16);
            acc[j] = __builtin_amdgcn_wmma_f32_16x16x32_bf16(false, af, false, bf,
                                                             (short)0, acc[j], false, false);
        }
    }
#else
    // fallback: cooperative LDS staging (128 threads x 32 B = 4 KB chunk)
    for (int ki = 0; ki < nk; ++ki) {
        __syncthreads();
        {
            int rr = tid >> 1, kk = (tid & 1) * 16;
            *(v16bf*)(&Bst[0][rr * 32 + kk]) =
                *(const v16bf*)(W + (size_t)(c0 + rr) * Kdim + (size_t)ki * 32 + kk);
        }
        __syncthreads();
        const __bf16* bbase = &Bst[0][0];
        v16bf af = load_a_frag(arow + ki * 32, hi);
#pragma unroll
        for (int j = 0; j < 4; ++j) {
            v16bf bf = *(const v16bf*)(bbase + (j * 16 + ln) * 32 + hi * 16);
            acc[j] = __builtin_amdgcn_wmma_f32_16x16x32_bf16(false, af, false, bf,
                                                             (short)0, acc[j], false, false);
        }
    }
#endif

#pragma unroll
    for (int j = 0; j < 4; ++j) {
        int col = c0 + j * 16 + ln;
        float bb = bias[col];
#pragma unroll
        for (int r = 0; r < 8; ++r) {
            int row = r0 + (hi ? 8 : 0) + r;
            float v = acc[j][r] + bb;
            if (EPI == 0) {
                outB[(size_t)row * N + col] = f2bf(v);
            } else if (EPI == 1) {
                float g = 0.5f * v * (1.0f + erff(v * 0.70710678118654752f));
                outB[(size_t)row * N + col] = f2bf(g);
            } else if (EPI == 2) {
                int sr = rowIdx[row];
                outF[(size_t)row * DIM + col] = v + gout[(size_t)sr * DIM + col];
            } else {
                float hv = v + resid[(size_t)row * DIM + col];
                if (!flag[part[row]])
                    gout[(size_t)rowIdx[row] * DIM + col] = hv;
            }
        }
    }
}

// ---------------- attention: one block (4 waves) per (m, head) ----------------
__global__ __launch_bounds__(128) void attn_kernel(const __bf16* __restrict__ qkv,
                                                   const float* __restrict__ maskF,
                                                   __bf16* __restrict__ attn_o) {
    const int mh = blockIdx.x;
    const int m  = mh / NHEAD;
    const int hh = mh % NHEAD;
    const int tid  = threadIdx.x;
    const int wave = tid >> 5;
    const int lane = tid & 31;
    const int ln   = lane & 15;
    const int hi   = lane >> 4;

    __shared__ __bf16 vT[DH][SEQ];            // v transposed: vT[d][key]
    __shared__ __bf16 probs[4][16][SEQ];      // per-wave softmax probs

    // stage v (64x32) transposed into LDS
    {
        int s  = tid & 63;
        int dg = tid >> 6;                    // 0..1 -> d chunks of 16
        const __bf16* vp = qkv + ((size_t)(m * SEQ + s)) * (3 * DIM) + hh * 96 + 64 + dg * 16;
        v8bf a = *(const v8bf*)(vp);
        v8bf b = *(const v8bf*)(vp + 8);
#pragma unroll
        for (int i = 0; i < 8; ++i) {
            vT[dg * 16 + i][s]     = a[i];
            vT[dg * 16 + 8 + i][s] = b[i];
        }
    }
    __syncthreads();

    const int s0 = wave * 16;                 // q-row strip for this wave

    // logits: Q (16x32) x K^T -> 16x64 strip, DH==32 == WMMA K -> one WMMA per tile
    const __bf16* qrow = qkv + ((size_t)(m * SEQ + s0 + ln)) * (3 * DIM) + hh * 96;
    v16bf afrag = load_a_frag(qrow, hi);
    v8f acc[4];
#pragma unroll
    for (int j = 0; j < 4; ++j) {
        const __bf16* krow = qkv + ((size_t)(m * SEQ + j * 16 + ln)) * (3 * DIM) + hh * 96 + DH;
        v16bf bfrag = *(const v16bf*)(krow + hi * 16);
        v8f z = {};
        acc[j] = __builtin_amdgcn_wmma_f32_16x16x32_bf16(false, afrag, false, bfrag,
                                                         (short)0, z, false, false);
    }

    // scale, mask (blocked key -> exactly -10000), row softmax over 64 keys
    const float scale = 0.17677669529663687f;   // 32^-0.5
    float l[4][8];
    float mx[8];
#pragma unroll
    for (int r = 0; r < 8; ++r) mx[r] = -1e30f;
#pragma unroll
    for (int j = 0; j < 4; ++j) {
        float mf = maskF[m * SEQ + j * 16 + ln];
#pragma unroll
        for (int r = 0; r < 8; ++r) {
            float v = (mf != 0.0f) ? -10000.0f : acc[j][r] * scale;
            l[j][r] = v;
            mx[r] = fmaxf(mx[r], v);
        }
    }
#pragma unroll
    for (int r = 0; r < 8; ++r) mx[r] = wred_max16(mx[r]);
    float sum[8];
#pragma unroll
    for (int r = 0; r < 8; ++r) sum[r] = 0.f;
#pragma unroll
    for (int j = 0; j < 4; ++j)
#pragma unroll
        for (int r = 0; r < 8; ++r) { l[j][r] = __expf(l[j][r] - mx[r]); sum[r] += l[j][r]; }
#pragma unroll
    for (int r = 0; r < 8; ++r) sum[r] = 1.0f / wred_sum16(sum[r]);

    // write probs strip (16x64 bf16) to LDS; same-wave LDS ops are in-order
#pragma unroll
    for (int j = 0; j < 4; ++j)
#pragma unroll
        for (int r = 0; r < 8; ++r)
            probs[wave][(hi ? 8 : 0) + r][j * 16 + ln] = f2bf(l[j][r] * sum[r]);

    // O = P (16x64) x V (64x32): 2 d-tiles x 2 k-steps
#pragma unroll
    for (int dt = 0; dt < 2; ++dt) {
        v8f o = {};
#pragma unroll
        for (int ks = 0; ks < 2; ++ks) {
            v16bf pa = load_a_frag(&probs[wave][ln][ks * 32], hi);
            v16bf pb = *(const v16bf*)(&vT[dt * 16 + ln][ks * 32 + hi * 16]);
            o = __builtin_amdgcn_wmma_f32_16x16x32_bf16(false, pa, false, pb,
                                                        (short)0, o, false, false);
        }
#pragma unroll
        for (int r = 0; r < 8; ++r) {
            int row = m * SEQ + s0 + (hi ? 8 : 0) + r;
            int col = hh * DH + dt * 16 + ln;
            attn_o[(size_t)row * DIM + col] = f2bf(o[r]);
        }
    }
}

// ---------------- launcher ----------------
extern "C" void kernel_launch(void* const* d_in, const int* in_sizes, int n_in,
                              void* d_out, int out_size, void* d_ws, size_t ws_size,
                              hipStream_t stream) {
    const float* x     = (const float*)d_in[0];
    const int*   iw    = (const int*)d_in[1];
    const int*   ip    = (const int*)d_in[2];
    const int*   bi    = (const int*)d_in[3];
    const float* ln1w  = (const float*)d_in[6];
    const float* ln1b  = (const float*)d_in[7];
    const float* qkvw  = (const float*)d_in[8];
    const float* qkvb  = (const float*)d_in[9];
    const float* projw = (const float*)d_in[10];
    const float* projb = (const float*)d_in[11];
    const float* ln2w  = (const float*)d_in[12];
    const float* ln2b  = (const float*)d_in[13];
    const float* fc1w  = (const float*)d_in[14];
    const float* fc1b  = (const float*)d_in[15];
    const float* fc2w  = (const float*)d_in[16];
    const float* fc2b  = (const float*)d_in[17];
    float* out = (float*)d_out;
    char*  ws  = (char*)d_ws;

    size_t off = 0;
    auto take = [&](size_t bytes) { size_t r = off; off = (off + bytes + 255) & ~(size_t)255; return r; };
    __bf16* wQKV  = (__bf16*)(ws + take((size_t)(3 * DIM) * DIM * 2));
    __bf16* wPROJ = (__bf16*)(ws + take((size_t)DIM * DIM * 2));
    __bf16* wFC1  = (__bf16*)(ws + take((size_t)MLPD * DIM * 2));
    __bf16* wFC2  = (__bf16*)(ws + take((size_t)DIM * MLPD * 2));
    unsigned char* flag = (unsigned char*)(ws + take(NTOK));
    float* maskF  = (float*)(ws + take((size_t)NTOK * 4));
    int*   rowIdx = (int*)(ws + take((size_t)NTOK * 4));
    __bf16* xp    = (__bf16*)(ws + take((size_t)NTOK * DIM * 2));
    __bf16* qkvB  = (__bf16*)(ws + take((size_t)NTOK * (3 * DIM) * 2));
    __bf16* attnO = (__bf16*)(ws + take((size_t)NTOK * DIM * 2));
    float*  hBuf  = (float*)(ws + take((size_t)NTOK * DIM * 4));
    __bf16* h2    = (__bf16*)(ws + take((size_t)NTOK * DIM * 2));
    __bf16* g1    = (__bf16*)(ws + take((size_t)NTOK * MLPD * 2));

    // 1. LN1 over all windows -> fp32 into d_out (correct result for inactive/blocked rows)
    ln_kernel<false><<<(NWIN * TT) / 8, 256, 0, stream>>>(x, ln1w, ln1b, out, nullptr, NWIN * TT);

    // 2. weights -> bf16
    cast_w_kernel<<<(3 * DIM * DIM + 255) / 256, 256, 0, stream>>>(qkvw, wQKV, 3 * DIM * DIM);
    cast_w_kernel<<<(DIM * DIM + 255) / 256, 256, 0, stream>>>(projw, wPROJ, DIM * DIM);
    cast_w_kernel<<<(MLPD * DIM + 255) / 256, 256, 0, stream>>>(fc1w, wFC1, MLPD * DIM);
    cast_w_kernel<<<(DIM * MLPD + 255) / 256, 256, 0, stream>>>(fc2w, wFC2, DIM * MLPD);

    // 3. blocked-token flag
    hipMemsetAsync(flag, 0, NTOK, stream);
    scatter_flag_kernel<<<(NBLOCK + 255) / 256, 256, 0, stream>>>(bi, flag);

    // 4. gather xn -> xp (bf16), build mask + row indices
    gather_kernel<<<(NTOK * DIM) / 256, 256, 0, stream>>>(out, iw, ip, flag, xp, maskF, rowIdx);

    // 5. QKV GEMM (98304 x 384) x (1152 x 384)^T -> bf16
    gemm_kernel<0><<<dim3(NTOK / 64, (3 * DIM) / 64), 128, 0, stream>>>(
        xp, wQKV, qkvb, 3 * DIM, DIM, qkvB, nullptr, nullptr, nullptr, nullptr, nullptr, nullptr);

    // 6. attention per (m, head)
    attn_kernel<<<WACT * NHEAD, 128, 0, stream>>>(qkvB, maskF, attnO);

    // 7. proj GEMM + xn shortcut -> h (fp32)
    gemm_kernel<2><<<dim3(NTOK / 64, DIM / 64), 128, 0, stream>>>(
        attnO, wPROJ, projb, DIM, DIM, nullptr, hBuf, nullptr, rowIdx, nullptr, nullptr, out);

    // 8. LN2 -> bf16
    ln_kernel<true><<<NTOK / 8, 256, 0, stream>>>(hBuf, ln2w, ln2b, nullptr, h2, NTOK);

    // 9. FC1 + exact GELU -> bf16
    gemm_kernel<1><<<dim3(NTOK / 64, MLPD / 64), 128, 0, stream>>>(
        h2, wFC1, fc1b, MLPD, DIM, g1, nullptr, nullptr, nullptr, nullptr, nullptr, nullptr);

    // 10. FC2 + h residual, scatter into d_out (skip blocked tokens -> they keep xn)
    gemm_kernel<3><<<dim3(NTOK / 64, DIM / 64), 128, 0, stream>>>(
        g1, wFC2, fc2b, DIM, MLPD, nullptr, nullptr, hBuf, rowIdx, flag, ip, out);
}
